// LSAttentionNonCausal_56032143343974
// MI455X (gfx1250) — matrix operand order repaired and checked by reading, not compiled, not run on hardware
//
#include <hip/hip_runtime.h>

#define DIMX 1024
#define HN   16
#define DHN  64
#define LC   32
#define WWIN 8
#define EEXT 4
#define SEQN 4096
#define BN   4
#define HLN  (HN * LC)     // 512
#define BANDN (WWIN + 2 * EEXT) // 16
#define LN_EPS 1e-5f

typedef unsigned short u16;
typedef unsigned int   u32;
typedef unsigned long long u64;
typedef __attribute__((ext_vector_type(16))) __bf16 v16bf;
typedef __attribute__((ext_vector_type(8)))  float  v8f;

union Frag {
  uint4 q[2];
  u32   d[8];
  v16bf v;
};

__device__ __forceinline__ v8f zero8() {
  v8f z = {0.f, 0.f, 0.f, 0.f, 0.f, 0.f, 0.f, 0.f};
  return z;
}

__device__ __forceinline__ u16 f2bf(float f) {
  union { float f; u32 u; } v; v.f = f;
  u32 u = v.u;
  return (u16)((u + 0x7fffu + ((u >> 16) & 1u)) >> 16);
}
__device__ __forceinline__ float bf2f(u16 h) {
  union { float f; u32 u; } v; v.u = ((u32)h) << 16;
  return v.f;
}

__device__ __forceinline__ v8f wmma_bf16(v16bf a, v16bf b, v8f c) {
  return __builtin_amdgcn_wmma_f32_16x16x32_bf16(false, a, false, b, (short)0, c, false, false);
}

// Async DMA global -> LDS (16B per lane), tracked by ASYNCcnt.
#define ASYNC_LOAD_B128(ldsoff, gaddr)                                        \
  asm volatile("global_load_async_to_lds_b128 %0, %1, off"                    \
               :: "v"(ldsoff), "v"(gaddr) : "memory")
#define ASYNC_WAIT_0() asm volatile("s_wait_asynccnt 0" ::: "memory")
#define ASYNC_WAIT_2() asm volatile("s_wait_asynccnt 2" ::: "memory")
#define ASYNC_WAIT_8() asm volatile("s_wait_asynccnt 8" ::: "memory")

__device__ __forceinline__ u32 lds_off(const void* p) {
  return (u32)(uintptr_t)p; // generic LDS pointer: addr[31:0] is the LDS offset
}

// ---------------------------------------------------------------------------
// query (S,B,DIM) f32 -> X (B,S,DIM) bf16
// ---------------------------------------------------------------------------
__global__ __launch_bounds__(256) void k_cvt_x(const float* __restrict__ q,
                                               u16* __restrict__ X) {
  size_t i = ((size_t)blockIdx.x * 256 + threadIdx.x) * 4; // flat (b,s,d)
  int d = (int)(i & (DIMX - 1));
  size_t t = i >> 10;
  int s = (int)(t & (SEQN - 1));
  int b = (int)(t >> 12);
  const float4 f = *(const float4*)&q[((size_t)s * BN + b) * DIMX + d];
  uint2 o;
  o.x = (u32)f2bf(f.x) | ((u32)f2bf(f.y) << 16);
  o.y = (u32)f2bf(f.z) | ((u32)f2bf(f.w) << 16);
  *(uint2*)&X[i] = o;
}

// ---------------------------------------------------------------------------
// Pack weight (K x N) f32 -> bf16 WMMA B-fragment layout:
// dword index = ((nt*KT + kt)*32 + lane)*8 + j ; lane: n = nt*16+(lane&15),
// khalf = lane>>4 ; k = kt*32 + khalf*16 + 2j (pair packed in dword)
// ---------------------------------------------------------------------------
__global__ __launch_bounds__(256) void k_pack_w(const float* __restrict__ Wsrc,
                                                u32* __restrict__ Wpk,
                                                int K, int N) {
  size_t i = (size_t)blockIdx.x * 256 + threadIdx.x;
  size_t total = (size_t)K * N / 2;
  if (i >= total) return;
  int j = (int)(i & 7);
  size_t t1 = i >> 3;
  int lane = (int)(t1 & 31);
  size_t t2 = t1 >> 5;
  int KT = K >> 5;
  int kt = (int)(t2 % KT);
  int nt = (int)(t2 / KT);
  int n = nt * 16 + (lane & 15);
  int k = kt * 32 + (lane >> 4) * 16 + 2 * j;
  u16 lo = f2bf(Wsrc[(size_t)k * N + n]);
  u16 hi = f2bf(Wsrc[(size_t)(k + 1) * N + n]);
  Wpk[i] = (u32)lo | ((u32)hi << 16);
}

// ---------------------------------------------------------------------------
// GEMM: out(MxN) = A(MxK,bf16) * Bpk(packed bf16) + bias, optional scale.
// MODE 0: f32 out ; MODE 1: bf16 out ; MODE 2: f32 out transposed (s*B+b)
// 8 waves, macro tile 128x64, wave tile 32x32, K step 32.
// A tiles staged with async DMA to LDS, triple-buffered: issue k+2 / compute k
// while tile k+1 is in flight (2 ASYNCcnt per wave per tile).
// ---------------------------------------------------------------------------
template <int MODE>
__global__ __launch_bounds__(256) void k_gemm(const u16* __restrict__ A,
                                              const u32* __restrict__ Bpk,
                                              const float* __restrict__ bias,
                                              float* __restrict__ outF,
                                              u16* __restrict__ outH,
                                              int M, int N, int K, float scale) {
  __shared__ u16 As[3][128][40]; // 80B stride: 16B aligned, conflict-free
  const int tid = threadIdx.x;
  const int lane = tid & 31, wv = tid >> 5;
  const int l15 = lane & 15, khalf = lane >> 4;
  const int m0 = blockIdx.y * 128;
  const int n0 = blockIdx.x * 64;
  const int KT = K >> 5;
  const int mw = (wv >> 1) * 32;

  v8f acc[2][2];
#pragma unroll
  for (int t = 0; t < 2; ++t)
#pragma unroll
    for (int n = 0; n < 2; ++n) acc[t][n] = zero8();

  // Each thread DMAs two 16B chunks per tile (units tid and tid+256).
  auto issueTile = [&](int kt, int buf) {
#pragma unroll
    for (int half = 0; half < 2; ++half) {
      int u = tid + half * 256;         // 512 units of 16B = 8KB tile
      int row = u >> 2, qq = u & 3;
      const u16* src = A + (size_t)(m0 + row) * K + kt * 32 + qq * 8;
      u32 dst = lds_off(&As[buf][row][qq * 8]);
      ASYNC_LOAD_B128(dst, (u64)(uintptr_t)src);
    }
  };

  issueTile(0, 0);
  issueTile(1, 1);
  ASYNC_WAIT_2(); // tile 0 resident (tile 1 may remain in flight)

  for (int kt = 0; kt < KT; ++kt) {
    __syncthreads(); // publish tile kt to all waves; all reads of buf (kt+2)%3 done
    if (kt + 2 < KT) issueTile(kt + 2, (kt + 2) % 3);
    const int buf = kt % 3;

    Frag a[2], bfr[2];
#pragma unroll
    for (int t = 0; t < 2; ++t) {
      const u16* r = &As[buf][mw + t * 16 + l15][0];
      a[t].q[0] = *(const uint4*)(r + khalf * 16);
      a[t].q[1] = *(const uint4*)(r + khalf * 16 + 8);
    }
#pragma unroll
    for (int n = 0; n < 2; ++n) {
      size_t ntg = (size_t)((n0 >> 4) + (wv & 1) * 2 + n);
      const uint4* bp = (const uint4*)Bpk + ((ntg * KT + kt) * 32 + lane) * 2;
      bfr[n].q[0] = bp[0];
      bfr[n].q[1] = bp[1];
    }
#pragma unroll
    for (int t = 0; t < 2; ++t)
#pragma unroll
      for (int n = 0; n < 2; ++n) acc[t][n] = wmma_bf16(a[t].v, bfr[n].v, acc[t][n]);

    if (kt + 1 < KT) {
      if (kt + 2 < KT) ASYNC_WAIT_2(); // tile kt+1 resident, kt+2 may fly
      else             ASYNC_WAIT_0(); // drain for the final tile
    }
  }

#pragma unroll
  for (int t = 0; t < 2; ++t) {
#pragma unroll
    for (int n = 0; n < 2; ++n) {
      int col = n0 + (wv & 1) * 32 + n * 16 + l15;
      float bv = bias ? bias[col] : 0.f;
#pragma unroll
      for (int r = 0; r < 8; ++r) {
        int row = m0 + mw + t * 16 + ((lane < 16) ? r : r + 8);
        float val = (acc[t][n][r] + bv) * scale;
        if (MODE == 0) {
          outF[(size_t)row * N + col] = val;
        } else if (MODE == 1) {
          outH[(size_t)row * N + col] = f2bf(val);
        } else {
          int bb = row >> 12;           // row = b*SEQ + s
          int ss = row & (SEQN - 1);
          outF[((size_t)ss * BN + bb) * N + col] = val;
        }
      }
    }
  }
}

// ---------------------------------------------------------------------------
// Row LayerNorm over DIM=1024, f32 in -> bf16 out
// ---------------------------------------------------------------------------
__global__ __launch_bounds__(256) void k_ln(const float* __restrict__ X,
                                            const float* __restrict__ g,
                                            const float* __restrict__ bt,
                                            u16* __restrict__ O) {
  __shared__ float red[256];
  const int row = blockIdx.x, tid = threadIdx.x;
  const float* xr = X + (size_t)row * DIMX;
  float x[4];
  float s = 0.f;
#pragma unroll
  for (int k = 0; k < 4; ++k) { x[k] = xr[tid + k * 256]; s += x[k]; }
  red[tid] = s; __syncthreads();
  for (int o = 128; o; o >>= 1) { if (tid < o) red[tid] += red[tid + o]; __syncthreads(); }
  float mean = red[0] * (1.f / DIMX);
  __syncthreads();
  s = 0.f;
#pragma unroll
  for (int k = 0; k < 4; ++k) { float d = x[k] - mean; s += d * d; }
  red[tid] = s; __syncthreads();
  for (int o = 128; o; o >>= 1) { if (tid < o) red[tid] += red[tid + o]; __syncthreads(); }
  float rstd = rsqrtf(red[0] * (1.f / DIMX) + LN_EPS);
  u16* orow = O + (size_t)row * DIMX;
#pragma unroll
  for (int k = 0; k < 4; ++k) {
    int i = tid + k * 256;
    orow[i] = f2bf((x[k] - mean) * rstd * g[i] + bt[i]);
  }
}

// ---------------------------------------------------------------------------
// head_scores = softmax over s (axis=1) of D (b,s,HL); write bf16 transposed
// into Dsm (b,h,l,s).  One block per (b, column).
// ---------------------------------------------------------------------------
__global__ __launch_bounds__(256) void k_softmax_seq(const float* __restrict__ D,
                                                     u16* __restrict__ Dsm) {
  __shared__ float red[256];
  const int bc = blockIdx.x;
  const int b = bc >> 9;
  const int c = bc & 511;
  const int h = c >> 5, lc = c & 31;
  const int tid = threadIdx.x;
  float v[16];
  float mx = -3.4e38f;
#pragma unroll
  for (int k = 0; k < 16; ++k) {
    int s = tid + k * 256;
    v[k] = D[((size_t)b * SEQN + s) * HLN + c];
    mx = fmaxf(mx, v[k]);
  }
  red[tid] = mx; __syncthreads();
  for (int o = 128; o; o >>= 1) { if (tid < o) red[tid] = fmaxf(red[tid], red[tid + o]); __syncthreads(); }
  mx = red[0]; __syncthreads();
  float sum = 0.f;
#pragma unroll
  for (int k = 0; k < 16; ++k) { v[k] = __expf(v[k] - mx); sum += v[k]; }
  red[tid] = sum; __syncthreads();
  for (int o = 128; o; o >>= 1) { if (tid < o) red[tid] += red[tid + o]; __syncthreads(); }
  float inv = 1.f / red[0];
  u16* dst = Dsm + (((size_t)b * HN + h) * LC + lc) * SEQN;
#pragma unroll
  for (int k = 0; k < 16; ++k) { int s = tid + k * 256; dst[s] = f2bf(v[k] * inv); }
}

// ---------------------------------------------------------------------------
// Kc/Vc = hs(L x S) @ K/V(S x DH) per (b,h) via WMMA. Block = 2 waves:
// wave0 -> Kc, wave1 -> Vc. Waves are fully independent: wave-private async
// double-buffered staging, no workgroup barriers.
// Output f32 in (b, l, h, dh) layout (for LN rows).
// ---------------------------------------------------------------------------
__global__ __launch_bounds__(64) void k_cmp_kv(const u16* __restrict__ Dsm,
                                               const u16* __restrict__ Kbf,
                                               const u16* __restrict__ Vbf,
                                               float* __restrict__ Kc,
                                               float* __restrict__ Vc) {
  __shared__ u16 Tile[2][2][32][72]; // [wave][buf][row][64+pad]
  const int bh = blockIdx.x;
  const int b = bh >> 4, h = bh & 15;
  const int tid = threadIdx.x, lane = tid & 31, wv = tid >> 5;
  const int l15 = lane & 15, khalf = lane >> 4;
  const u16* src = wv ? Vbf : Kbf;
  float* dst = wv ? Vc : Kc;
  const u16* Arow0 = Dsm + ((size_t)b * HN + h) * LC * SEQN;

  v8f acc[2][4];
#pragma unroll
  for (int t = 0; t < 2; ++t)
#pragma unroll
    for (int n = 0; n < 4; ++n) acc[t][n] = zero8();

  // lane stages one 128B row of the 32x64 tile: 8 async b128 per lane.
  auto issueT = [&](int kt, int buf) {
    const u16* srow = src + ((size_t)b * SEQN + kt * 32 + lane) * DIMX + h * DHN;
    u32 l0 = lds_off(&Tile[wv][buf][lane][0]);
#pragma unroll
    for (int j = 0; j < 8; ++j)
      ASYNC_LOAD_B128(l0 + j * 16, (u64)(uintptr_t)(srow + j * 8));
  };

  const int NK = SEQN / 32;
  issueT(0, 0);
  for (int kt = 0; kt < NK; ++kt) {
    if (kt + 1 < NK) { issueT(kt + 1, (kt + 1) & 1); ASYNC_WAIT_8(); }
    else             { ASYNC_WAIT_0(); }
    const int buf = kt & 1;
    const int sb = kt * 32;

    Frag a[2];
#pragma unroll
    for (int t = 0; t < 2; ++t) {
      const u16* r = Arow0 + (size_t)(t * 16 + l15) * SEQN + sb;
      a[t].q[0] = *(const uint4*)(r + khalf * 16);
      a[t].q[1] = *(const uint4*)(r + khalf * 16 + 8);
    }
#pragma unroll
    for (int n = 0; n < 4; ++n) {
      Frag bfr;
#pragma unroll
      for (int j = 0; j < 8; ++j) {
        int k = khalf * 16 + 2 * j;
        bfr.d[j] = (u32)Tile[wv][buf][k][n * 16 + l15] |
                   ((u32)Tile[wv][buf][k + 1][n * 16 + l15] << 16);
      }
#pragma unroll
      for (int t = 0; t < 2; ++t) acc[t][n] = wmma_bf16(a[t].v, bfr.v, acc[t][n]);
    }
  }

#pragma unroll
  for (int t = 0; t < 2; ++t)
#pragma unroll
    for (int n = 0; n < 4; ++n)
#pragma unroll
      for (int r = 0; r < 8; ++r) {
        int lr = t * 16 + ((lane < 16) ? r : r + 8);
        int dh = n * 16 + l15;
        dst[(((size_t)b * LC + lr) * HN + h) * DHN + dh] = acc[t][n][r];
      }
}

// ---------------------------------------------------------------------------
// Fused attention per (b, h, 32 rows): compress scores (WMMA) + window scores
// (VALU) -> joint softmax(48) -> compress PV (WMMA) + window PV (VALU).
// Output C bf16 in (b,s,DIM) for the final Wo GEMM.
// ---------------------------------------------------------------------------
__global__ __launch_bounds__(64) void k_attn(const u16* __restrict__ Qbf,
                                             const u16* __restrict__ Kbf,
                                             const u16* __restrict__ Vbf,
                                             const u16* __restrict__ Kcn,
                                             const u16* __restrict__ Vcn,
                                             u16* __restrict__ Cbf) {
  __shared__ float sc[32][64];
  __shared__ float cw[32][64];
  __shared__ u16 Pc[32][40];
  __shared__ u16 Vt[32][72];
  const int blk = blockIdx.x;
  const int sblk = blk & 127;
  const int h = (blk >> 7) & 15;
  const int b = blk >> 11;
  const int s0 = sblk * 32;
  const int tid = threadIdx.x, lane = tid & 31, wv = tid >> 5;
  const int l15 = lane & 15, khalf = lane >> 4;

  { // stage Vcn 32x64 tile
    int row = tid >> 1, half = tid & 1;
    const uint4* s4 = (const uint4*)(Vcn + (((size_t)b * LC + row) * HN + h) * DHN + half * 32);
    uint4* d4 = (uint4*)&Vt[row][half * 32];
    d4[0] = s4[0]; d4[1] = s4[1];
  }

  // --- compress scores: rows s0+wv*16.. x Kcn^T (M=16,N=32,K=64) ---
  v8f accS[2] = {zero8(), zero8()};
#pragma unroll
  for (int ks = 0; ks < 2; ++ks) {
    Frag a;
    const u16* qr = Qbf + ((size_t)b * SEQN + s0 + wv * 16 + l15) * DIMX + h * DHN + ks * 32;
    a.q[0] = *(const uint4*)(qr + khalf * 16);
    a.q[1] = *(const uint4*)(qr + khalf * 16 + 8);
#pragma unroll
    for (int n = 0; n < 2; ++n) {
      Frag bfr;
      const u16* kr = Kcn + (((size_t)b * LC + n * 16 + l15) * HN + h) * DHN + ks * 32 + khalf * 16;
      bfr.q[0] = *(const uint4*)(kr);
      bfr.q[1] = *(const uint4*)(kr + 8);
      accS[n] = wmma_bf16(a.v, bfr.v, accS[n]);
    }
  }
#pragma unroll
  for (int n = 0; n < 2; ++n)
#pragma unroll
    for (int r = 0; r < 8; ++r) {
      int rl = wv * 16 + ((lane < 16) ? r : r + 8);
      sc[rl][n * 16 + l15] = accS[n][r];
    }
  __syncthreads();

  // --- window scores (band=16), -inf mask for invalid positions ---
  for (int idx = tid; idx < 512; idx += 64) {
    int r = idx >> 4, c = idx & 15;
    int s = s0 + r;
    int pos = (s / WWIN) * WWIN + c - EEXT;
    float val = -3.4e38f;
    if (pos >= 0 && pos < SEQN) {
      const u16* qr = Qbf + ((size_t)b * SEQN + s) * DIMX + h * DHN;
      const u16* kr = Kbf + ((size_t)b * SEQN + pos) * DIMX + h * DHN;
      float acc = 0.f;
#pragma unroll 8
      for (int d = 0; d < DHN; ++d) acc += bf2f(qr[d]) * bf2f(kr[d]);
      val = acc;
    }
    sc[r][32 + c] = val;
  }
  __syncthreads();

  // --- joint softmax over 48 ---
  if (tid < 32) {
    int r = tid;
    float m = -3.4e38f;
#pragma unroll
    for (int c = 0; c < 48; ++c) m = fmaxf(m, sc[r][c]);
    float e[48];
    float sum = 0.f;
#pragma unroll
    for (int c = 0; c < 48; ++c) {
      float x = sc[r][c];
      float ev = (x < -1e37f) ? 0.f : __expf(x - m);
      e[c] = ev; sum += ev;
    }
    float inv = 1.f / sum;
#pragma unroll
    for (int c = 0; c < 32; ++c) Pc[r][c] = f2bf(e[c] * inv);
#pragma unroll
    for (int c = 0; c < 16; ++c) sc[r][32 + c] = e[32 + c] * inv;
  }
  __syncthreads();

  // --- compress PV: P(16x32) @ Vc(32x64), single K step ---
  v8f accC[4] = {zero8(), zero8(), zero8(), zero8()};
  {
    Frag a;
    const u16* pr = &Pc[wv * 16 + l15][0];
    a.q[0] = *(const uint4*)(pr + khalf * 16);
    a.q[1] = *(const uint4*)(pr + khalf * 16 + 8);
#pragma unroll
    for (int n = 0; n < 4; ++n) {
      Frag bfr;
#pragma unroll
      for (int j = 0; j < 8; ++j) {
        int k = khalf * 16 + 2 * j;
        bfr.d[j] = (u32)Vt[k][n * 16 + l15] | ((u32)Vt[k + 1][n * 16 + l15] << 16);
      }
      accC[n] = wmma_bf16(a.v, bfr.v, accC[n]);
    }
  }

  // --- window PV (VALU): thread -> (row, dh half) ---
  {
    int r = tid & 31, hh = tid >> 5;
    float accv[32];
#pragma unroll
    for (int d = 0; d < 32; ++d) accv[d] = 0.f;
    int s = s0 + r;
    int g0 = (s / WWIN) * WWIN;
    for (int c = 0; c < BANDN; ++c) {
      float p = sc[r][32 + c];
      int pos = g0 + c - EEXT;
      if (p != 0.f && pos >= 0 && pos < SEQN) {
        const u16* vr = Vbf + ((size_t)b * SEQN + pos) * DIMX + h * DHN + hh * 32;
#pragma unroll
        for (int d = 0; d < 32; ++d) accv[d] += p * bf2f(vr[d]);
      }
    }
#pragma unroll
    for (int d = 0; d < 32; ++d) cw[r][hh * 32 + d] = accv[d];
  }
  __syncthreads();

  // --- combine and store bf16 C (b,s,DIM) ---
#pragma unroll
  for (int n = 0; n < 4; ++n)
#pragma unroll
    for (int r = 0; r < 8; ++r) {
      int rl = wv * 16 + ((lane < 16) ? r : r + 8);
      int dh = n * 16 + l15;
      float val = accC[n][r] + cw[rl][dh];
      Cbf[((size_t)b * SEQN + s0 + rl) * DIMX + h * DHN + dh] = f2bf(val);
    }
}

// ---------------------------------------------------------------------------
extern "C" void kernel_launch(void* const* d_in, const int* in_sizes, int n_in,
                              void* d_out, int out_size, void* d_ws, size_t ws_size,
                              hipStream_t stream) {
  const float* query = (const float*)d_in[0];
  const float* Wq = (const float*)d_in[1];  const float* bq = (const float*)d_in[2];
  const float* Wk = (const float*)d_in[3];  const float* bk = (const float*)d_in[4];
  const float* Wv = (const float*)d_in[5];  const float* bv = (const float*)d_in[6];
  const float* Wo = (const float*)d_in[7];  const float* bo = (const float*)d_in[8];
  const float* g_l = (const float*)d_in[9]; const float* b_l = (const float*)d_in[10];
  const float* g_s = (const float*)d_in[11];const float* b_s = (const float*)d_in[12];
  const float* Wd = (const float*)d_in[13]; const float* bd = (const float*)d_in[14];

  char* p = (char*)d_ws;
  auto alloc = [&](size_t bytes) -> void* {
    void* r = (void*)p;
    p += (bytes + 255) & ~(size_t)255;
    return r;
  };
  const size_t BS = (size_t)BN * SEQN;          // 16384 rows
  u16* Xbf  = (u16*)alloc(BS * DIMX * 2);
  u32* Wqpk = (u32*)alloc((size_t)DIMX * DIMX * 2);
  u32* Wkpk = (u32*)alloc((size_t)DIMX * DIMX * 2);
  u32* Wvpk = (u32*)alloc((size_t)DIMX * DIMX * 2);
  u32* Wopk = (u32*)alloc((size_t)DIMX * DIMX * 2);
  u32* Wdpk = (u32*)alloc((size_t)DIMX * HLN * 2);
  u16* Qbf  = (u16*)alloc(BS * DIMX * 2);
  u16* Kbf  = (u16*)alloc(BS * DIMX * 2);
  u16* Vbf  = (u16*)alloc(BS * DIMX * 2);
  u16* Cbf  = (u16*)alloc(BS * DIMX * 2);
  float* TMP = (float*)alloc(BS * DIMX * 4);
  u16* Dsm  = (u16*)alloc((size_t)BN * HN * LC * SEQN * 2);
  float* Kc = (float*)alloc((size_t)BN * LC * DIMX * 4);
  float* Vc = (float*)alloc((size_t)BN * LC * DIMX * 4);
  u16* Kcn  = (u16*)alloc((size_t)BN * LC * DIMX * 2);
  u16* Vcn  = (u16*)alloc((size_t)BN * LC * DIMX * 2);

  // 1) convert X, pack weights
  k_cvt_x<<<16384, 256, 0, stream>>>(query, Xbf);
  k_pack_w<<<2048, 256, 0, stream>>>(Wq, Wqpk, DIMX, DIMX);
  k_pack_w<<<2048, 256, 0, stream>>>(Wk, Wkpk, DIMX, DIMX);
  k_pack_w<<<2048, 256, 0, stream>>>(Wv, Wvpk, DIMX, DIMX);
  k_pack_w<<<2048, 256, 0, stream>>>(Wo, Wopk, DIMX, DIMX);
  k_pack_w<<<1024, 256, 0, stream>>>(Wd, Wdpk, DIMX, HLN);

  dim3 g1(DIMX / 64, (unsigned)(BS / 128));   // (16,128)
  dim3 gd(HLN / 64, (unsigned)(BS / 128));    // (8,128)

  // 2) projections
  k_gemm<1><<<g1, 256, 0, stream>>>(Xbf, Wqpk, bq, nullptr, Qbf, (int)BS, DIMX, DIMX, 0.125f);
  k_gemm<0><<<g1, 256, 0, stream>>>(Xbf, Wkpk, bk, TMP, nullptr, (int)BS, DIMX, DIMX, 1.f);
  k_ln<<<(unsigned)BS, 256, 0, stream>>>(TMP, g_l, b_l, Kbf);
  k_gemm<0><<<g1, 256, 0, stream>>>(Xbf, Wvpk, bv, TMP, nullptr, (int)BS, DIMX, DIMX, 1.f);
  k_ln<<<(unsigned)BS, 256, 0, stream>>>(TMP, g_l, b_l, Vbf);
  k_gemm<0><<<gd, 256, 0, stream>>>(Xbf, Wdpk, bd, TMP, nullptr, (int)BS, HLN, DIMX, 1.f);

  // 3) head-score softmax over sequence, transposed bf16 output
  k_softmax_seq<<<BN * HLN, 256, 0, stream>>>(TMP, Dsm);

  // 4) compressed K/V + their LayerNorm
  k_cmp_kv<<<BN * HN, 64, 0, stream>>>(Dsm, Kbf, Vbf, Kc, Vc);
  k_ln<<<BN * LC, 256, 0, stream>>>(Kc, g_s, b_s, Kcn);
  k_ln<<<BN * LC, 256, 0, stream>>>(Vc, g_s, b_s, Vcn);

  // 5) fused attention
  k_attn<<<BN * HN * (SEQN / 32), 64, 0, stream>>>(Qbf, Kbf, Vbf, Kcn, Vcn, Cbf);

  // 6) output projection with (s,b) transpose into d_out
  k_gemm<2><<<g1, 256, 0, stream>>>(Cbf, Wopk, bo, (float*)d_out, nullptr, (int)BS, DIMX, DIMX, 1.f);
}
